// PaperClassifier_3839700763100
// MI455X (gfx1250) — matrix operand order repaired and checked by reading, not compiled, not run on hardware
//
#include <hip/hip_runtime.h>

// ---------------------------------------------------------------------------
// PaperClassifier for MI455X (gfx1250): lm-proj + GRU scan + classifier head
// B=8, S=4096, D_IN=2048, D_H=512, N_CLS=10
//   K0 prep:   convert weights f32->bf16, split gate weights into x-part/h-part
//   K1 gemm1:  seq_bf16[32768,512] = x @ W_lm^T
//              (async global->LDS double-buffered A strip, WMMA bf16, 2 acc)
//   K2 gemm2:  A_bf16[3,8,4096,512] = seq @ W_{z,r,h}x^T
//              (async global->LDS single-shot A strip, WMMA bf16, 2 acc)
//   K3 scan:   16-WG cluster, weights resident in VGPRs, cluster barriers,
//              scoped release/acquire fences, split-K dual WMMA chains
//   K4 head:   out[8,10] = h_final @ W_c^T               (scalar f32)
// ---------------------------------------------------------------------------

typedef __attribute__((ext_vector_type(16))) __bf16 v16bf;
typedef __attribute__((ext_vector_type(8)))  __bf16 v8bf;
typedef __attribute__((ext_vector_type(8)))  float  v8f;

#define BB     8
#define SS     4096
#define DIN    2048
#define DH     512
#define MROWS  16          // padded batch rows for one WMMA M-tile
#define KCH    128         // GEMM1 K-chunk staged per LDS buffer

static __device__ __forceinline__ v8f wmma_bf16(v16bf a, v16bf b, v8f c) {
  return __builtin_amdgcn_wmma_f32_16x16x32_bf16(false, a, false, b,
                                                 (short)0, c, false, false);
}

// Async global->LDS 16B copy (ASYNCcnt-tracked; inline asm is portable across
// both toolchains per the bridge doc). lds_off = wave-relative LDS byte addr
// (low 32 bits of a generic pointer to __shared__).
static __device__ __forceinline__ void async_copy_b128(unsigned lds_off,
                                                       const void* gaddr) {
  asm volatile("global_load_async_to_lds_b128 %0, %1, off"
               :: "v"(lds_off), "v"(gaddr) : "memory");
}
static __device__ __forceinline__ void wait_async0() {
  asm volatile("s_wait_asynccnt 0" ::: "memory");
}

// A-fragment (16x32 bf16, ISA 7.12.2): lane r (0..15) row M=r; hi half of wave
// shifts K by 8 within each 16-K group. Two 16B contiguous loads.
static __device__ __forceinline__ v16bf load_afrag_bf16(const __bf16* base, int ld,
                                                        int r, int hi, int k0) {
  const __bf16* p0 = base + (size_t)r * ld + k0 + hi * 8;
  v8bf lo = *(const v8bf*)p0;
  v8bf hh = *(const v8bf*)(p0 + 16);
  v16bf out;
#pragma unroll
  for (int i = 0; i < 8; ++i) { out[i] = lo[i]; out[i + 8] = hh[i]; }
  return out;
}

// Same fragment shape sourced from f32 data (global or LDS), convert to bf16.
static __device__ __forceinline__ v16bf load_afrag_f32(const float* base, int ld,
                                                       int r, int hi, int k0) {
  const float* p0 = base + (size_t)r * ld + k0 + hi * 8;
  const float* p1 = p0 + 16;
  v16bf out;
#pragma unroll
  for (int i = 0; i < 8; ++i) { out[i] = (__bf16)p0[i]; out[i + 8] = (__bf16)p1[i]; }
  return out;
}

// B-fragment (32x16 bf16): lane r holds column N=r (row r of row-major W),
// hi half of wave shifts K by 8 within each 16-K group. Two 16B contiguous loads.
static __device__ __forceinline__ v16bf load_bfrag(const __bf16* wrow, int hi, int k0) {
  const __bf16* p0 = wrow + k0 + hi * 8;
  v8bf lo = *(const v8bf*)p0;
  v8bf hh = *(const v8bf*)(p0 + 16);
  v16bf out;
#pragma unroll
  for (int i = 0; i < 8; ++i) { out[i] = lo[i]; out[i + 8] = hh[i]; }
  return out;
}

static __device__ __forceinline__ void fence_rel_agent() {
  __builtin_amdgcn_fence(__ATOMIC_RELEASE, "agent");
}
static __device__ __forceinline__ void fence_acq_agent() {
  __builtin_amdgcn_fence(__ATOMIC_ACQUIRE, "agent");
}

// ---------------------------------------------------------------------------
// K0: weight conversion / splitting + state init
// ---------------------------------------------------------------------------
__global__ void pc_prep(const float* __restrict__ Wlm, const float* __restrict__ Wz,
                        const float* __restrict__ Wr, const float* __restrict__ Wh,
                        __bf16* __restrict__ wlm_bf, __bf16* __restrict__ wgx_bf,
                        __bf16* __restrict__ wgh_bf, __bf16* __restrict__ hglob,
                        __bf16* __restrict__ rhglob) {
  size_t tid = (size_t)blockIdx.x * blockDim.x + threadIdx.x;
  size_t stride = (size_t)gridDim.x * blockDim.x;
  for (size_t i = tid; i < (size_t)DH * DIN; i += stride)
    wlm_bf[i] = (__bf16)Wlm[i];
  for (size_t i = tid; i < (size_t)3 * DH * DH; i += stride) {
    int g = (int)(i / (DH * DH));
    int rem = (int)(i % (DH * DH));
    int n = rem / DH, k = rem % DH;
    const float* Wg = (g == 0) ? Wz : (g == 1) ? Wr : Wh;
    wgx_bf[i] = (__bf16)Wg[(size_t)n * (2 * DH) + k];         // x-part
    wgh_bf[i] = (__bf16)Wg[(size_t)n * (2 * DH) + DH + k];    // h-part
  }
  for (size_t i = tid; i < (size_t)2 * MROWS * DH; i += stride) hglob[i] = (__bf16)0.f;
  for (size_t i = tid; i < (size_t)MROWS * DH; i += stride)     rhglob[i] = (__bf16)0.f;
}

// ---------------------------------------------------------------------------
// K1: seq[32768,512] = x[32768,2048] @ Wlm^T
// 2 blocks per M-tile; 8 waves/block share one async-staged, double-buffered
// 16xKCH f32 A strip in LDS (+4-float row pad -> conflict-free bank mapping).
// Each wave: two N-tiles, shared A fragment, dual accumulators.
// ---------------------------------------------------------------------------
__global__ __launch_bounds__(256) void pc_gemm1(const float* __restrict__ x,
                                                const __bf16* __restrict__ wlm,
                                                __bf16* __restrict__ seq) {
  __shared__ float xs[2][16][KCH + 4];
  const int tid = threadIdx.x;
  const int wavei = tid >> 5, lane = tid & 31, r = lane & 15, hi = lane >> 4;
  const int mtile = blockIdx.x >> 1;               // 2048 M-tiles
  const int npair = (blockIdx.x & 1) * 8 + wavei;  // 16 N-tile pairs
  const float* A = x + (size_t)mtile * 16 * DIN;
  const __bf16* B0 = wlm + (size_t)(npair * 32 + r) * DIN;
  const __bf16* B1 = B0 + (size_t)16 * DIN;

  // stage(kc, buf): 16 rows x KCH f32 = 512 x 16B chunks, 2 per thread
  auto stage = [&](int kc, int buf) {
#pragma unroll
    for (int i = tid; i < 16 * (KCH / 4); i += 256) {
      int row = i >> 5;          // KCH/4 = 32 float4 per row
      int c4  = i & 31;
      async_copy_b128((unsigned)(uintptr_t)&xs[buf][row][c4 * 4],
                      A + (size_t)row * DIN + kc * KCH + c4 * 4);
    }
  };

  v8f acc0 = {}, acc1 = {};
  stage(0, 0);
  for (int kc = 0; kc < DIN / KCH; ++kc) {         // 16 chunks
    const int cur = kc & 1;
    wait_async0();                                 // my async copies landed
    __syncthreads();                               // everyone's copies landed
    if (kc + 1 < DIN / KCH) stage(kc + 1, cur ^ 1);
#pragma unroll
    for (int ks = 0; ks < KCH / 32; ++ks) {
      int kg = kc * KCH + ks * 32;
      v16bf a  = load_afrag_f32(&xs[cur][0][0], KCH + 4, r, hi, ks * 32);
      v16bf b0 = load_bfrag(B0, hi, kg);
      v16bf b1 = load_bfrag(B1, hi, kg);
      acc0 = wmma_bf16(a, b0, acc0);
      acc1 = wmma_bf16(a, b1, acc1);
    }
  }
#pragma unroll
  for (int v = 0; v < 8; ++v) {
    int M = mtile * 16 + v + 8 * hi;
    seq[(size_t)M * DH + npair * 32 + r]      = (__bf16)acc0[v];
    seq[(size_t)M * DH + npair * 32 + 16 + r] = (__bf16)acc1[v];
  }
}

// ---------------------------------------------------------------------------
// K2: A[g][b][t][n] = seq @ Wgx^T  (gate preactivation contributions from x)
// 6 blocks per M-tile; whole 16x512 bf16 A strip async-staged to LDS once.
// Each wave: two adjacent N-tiles (never straddles a gate: 32 tiles/gate).
// ---------------------------------------------------------------------------
__global__ __launch_bounds__(256) void pc_gemm2(const __bf16* __restrict__ seq,
                                                const __bf16* __restrict__ wgx,
                                                __bf16* __restrict__ Aout) {
  __shared__ __bf16 ss[16][DH + 8];
  const int tid = threadIdx.x;
  const int wavei = tid >> 5, lane = tid & 31, r = lane & 15, hi = lane >> 4;
  const int mtile = blockIdx.x / 6;                // 2048 M-tiles
  const int np    = (blockIdx.x % 6) * 8 + wavei;  // 48 pairs over 96 N-tiles
  const int ntile = np * 2;
  const int g  = ntile >> 5;
  const int n0 = (ntile & 31) * 16;
  const __bf16* A  = seq + (size_t)mtile * 16 * DH;
  const __bf16* B0 = wgx + ((size_t)g * DH + n0 + r) * DH;
  const __bf16* B1 = B0 + (size_t)16 * DH;

  // stage strip: 16 rows x 512 bf16 = 1024 x 16B chunks, 4 per thread
#pragma unroll
  for (int i = tid; i < 16 * (DH / 8); i += 256) {
    int row = i >> 6;            // DH/8 = 64 x 16B chunks per row
    int c   = i & 63;
    async_copy_b128((unsigned)(uintptr_t)&ss[row][c * 8],
                    A + (size_t)row * DH + c * 8);
  }
  wait_async0();
  __syncthreads();

  v8f acc0 = {}, acc1 = {};
#pragma unroll 4
  for (int kc = 0; kc < DH / 32; ++kc) {
    int k0 = kc * 32;
    v16bf a  = load_afrag_bf16(&ss[0][0], DH + 8, r, hi, k0);
    v16bf b0 = load_bfrag(B0, hi, k0);
    v16bf b1 = load_bfrag(B1, hi, k0);
    acc0 = wmma_bf16(a, b0, acc0);
    acc1 = wmma_bf16(a, b1, acc1);
  }
#pragma unroll
  for (int v = 0; v < 8; ++v) {
    int M = mtile * 16 + v + 8 * hi;   // seq row = b*S + t
    int bb = M >> 12, t = M & (SS - 1);
    size_t base = (((size_t)g * BB + bb) * SS + t) * DH;
    Aout[base + n0 + r]      = (__bf16)acc0[v];
    Aout[base + n0 + 16 + r] = (__bf16)acc1[v];
  }
}

// ---------------------------------------------------------------------------
// K3: GRU scan. 16 workgroups (one cluster), 128 threads (4 waves) each.
// WG w owns output columns [w*32, w*32+32). Wave (gate,nt): gate 0=z wave
// keeps z in regs and finalizes h; gate 1=r wave produces rh. W_*h slices
// live in VGPRs for the whole scan. Split-K dual WMMA chains for ILP.
// h/rh exchanged through L2 with release/acquire fences + cluster barriers.
// ---------------------------------------------------------------------------
__global__ __launch_bounds__(128, 1) void pc_scan(const __bf16* __restrict__ wgh,
                                                  const __bf16* __restrict__ Ain,
                                                  __bf16* hglob, __bf16* rhglob) {
  const int wg   = blockIdx.x;          // 0..15
  const int wave = threadIdx.x >> 5;    // 0..3
  const int lane = threadIdx.x & 31;
  const int r    = lane & 15, hi = lane >> 4;
  const int gate = wave >> 1;           // 0 = z, 1 = r
  const int nt   = wave & 1;            // N-tile within this WG's 32 columns
  const int mycol = wg * 32 + nt * 16 + r;

  __shared__ float partial[2][8][32];   // phase-B K-half reduction staging

  // Persistent weight fragments in VGPRs (loaded once, never reloaded):
  v16bf wB[16];                         // own gate's W_gh columns, full K=512
  const __bf16* Wg = wgh + (size_t)gate * DH * DH + (size_t)mycol * DH;
#pragma unroll
  for (int kc = 0; kc < 16; ++kc) wB[kc] = load_bfrag(Wg, hi, kc * 32);
  v16bf hB[8];                          // W_hh, K-half selected by gate
  const __bf16* Wh2 = wgh + (size_t)2 * DH * DH + (size_t)mycol * DH + gate * 256;
#pragma unroll
  for (int kc = 0; kc < 8; ++kc) hB[kc] = load_bfrag(Wh2, hi, kc * 32);

  for (int t = 0; t < SS; ++t) {
    const __bf16* hcur  = hglob + (size_t)(t & 1) * MROWS * DH;
    __bf16*       hnext = hglob + (size_t)((t & 1) ^ 1) * MROWS * DH;

    // This step's x-side preactivations (issued early), plus prefetch of the
    // next step's lines so they are L2/WGP$-warm when step t+1 starts.
    float apre[8];
#pragma unroll
    for (int v = 0; v < 8; ++v) {
      int M = v + 8 * hi;
      if (M < BB) {
        const __bf16* p = &Ain[(((size_t)gate * BB + M) * SS + t) * DH + mycol];
        apre[v] = (float)*p;
        __builtin_prefetch(p + DH, 0, 1);          // same (g,b), t+1
      } else {
        apre[v] = 0.f;
      }
    }

    // ---- phase A: gate preactivation, two independent WMMA chains ----
    v8f accA = {}, accB = {};
#pragma unroll
    for (int kc = 0; kc < 8; ++kc) {
      v16bf a0 = load_afrag_bf16(hcur, DH, r, hi, kc * 64);
      v16bf a1 = load_afrag_bf16(hcur, DH, r, hi, kc * 64 + 32);
      accA = wmma_bf16(a0, wB[2 * kc],     accA);
      accB = wmma_bf16(a1, wB[2 * kc + 1], accB);
    }
    float gval[8];
#pragma unroll
    for (int v = 0; v < 8; ++v) {
      float pre = accA[v] + accB[v] + apre[v];
      gval[v] = 1.f / (1.f + __expf(-pre));
    }
    if (gate == 1) {                    // r-wave: publish r*h
#pragma unroll
      for (int v = 0; v < 8; ++v) {
        int M = v + 8 * hi;
        float hold = (float)hcur[(size_t)M * DH + mycol];
        rhglob[(size_t)M * DH + mycol] = (__bf16)(gval[v] * hold);
      }
    }
    fence_rel_agent();
    __syncthreads();
    __builtin_amdgcn_s_cluster_barrier();   // s_barrier_signal/wait -3
    fence_acq_agent();

    // ---- phase B: candidate h' (each wave one K-half of one N-tile) ----
    float ahv[8], holdv[8];
#pragma unroll
    for (int v = 0; v < 8; ++v) {           // early loads to overlap WMMA
      int M = v + 8 * hi;
      ahv[v] = (gate == 0 && M < BB)
          ? (float)Ain[(((size_t)2 * BB + M) * SS + t) * DH + mycol] : 0.f;
      holdv[v] = (float)hcur[(size_t)M * DH + mycol];
    }
    v8f acc2A = {}, acc2B = {};
#pragma unroll
    for (int kc = 0; kc < 4; ++kc) {
      int kb = gate * 256 + kc * 64;
      v16bf a0 = load_afrag_bf16(rhglob, DH, r, hi, kb);
      v16bf a1 = load_afrag_bf16(rhglob, DH, r, hi, kb + 32);
      acc2A = wmma_bf16(a0, hB[2 * kc],     acc2A);
      acc2B = wmma_bf16(a1, hB[2 * kc + 1], acc2B);
    }
    if (gate == 1) {
#pragma unroll
      for (int v = 0; v < 8; ++v) partial[nt][v][lane] = acc2A[v] + acc2B[v];
    }
    __syncthreads();
    if (gate == 0) {                    // z-wave finalizes (it holds z)
#pragma unroll
      for (int v = 0; v < 8; ++v) {
        float pre = acc2A[v] + acc2B[v] + partial[nt][v][lane] + ahv[v];
        float hp  = tanhf(pre);
        float z   = gval[v];
        int M = v + 8 * hi;
        hnext[(size_t)M * DH + mycol] = (__bf16)((1.f - z) * holdv[v] + z * hp);
      }
    }
    fence_rel_agent();
    __syncthreads();
    __builtin_amdgcn_s_cluster_barrier();
    fence_acq_agent();
  }
}

// ---------------------------------------------------------------------------
// K4: classifier head  out[8,10] = h_final @ Wc^T  (h_final lives in hglob[0])
// ---------------------------------------------------------------------------
__global__ void pc_head(const __bf16* __restrict__ h, const float* __restrict__ Wc,
                        float* __restrict__ out) {
  int i = threadIdx.x;
  if (i < BB * 10) {
    int b = i / 10, c = i % 10;
    float s = 0.f;
    for (int n = 0; n < DH; ++n)
      s += (float)h[(size_t)b * DH + n] * Wc[(size_t)c * DH + n];
    out[i] = s;
  }
}

// ---------------------------------------------------------------------------
extern "C" void kernel_launch(void* const* d_in, const int* in_sizes, int n_in,
                              void* d_out, int out_size, void* d_ws, size_t ws_size,
                              hipStream_t stream) {
  const float* x   = (const float*)d_in[0];
  const float* Wlm = (const float*)d_in[1];
  const float* Wz  = (const float*)d_in[2];
  const float* Wr  = (const float*)d_in[3];
  const float* Wh  = (const float*)d_in[4];
  const float* Wc  = (const float*)d_in[5];
  float* out = (float*)d_out;

  char* ws = (char*)d_ws;
  size_t o = 0;
  auto alloc = [&](size_t bytes) { char* p = ws + o; o = (o + bytes + 255) & ~(size_t)255; return p; };
  __bf16* wlm_bf = (__bf16*)alloc((size_t)DH * DIN * 2);          //   2 MB
  __bf16* wgx_bf = (__bf16*)alloc((size_t)3 * DH * DH * 2);       // 1.5 MB
  __bf16* wgh_bf = (__bf16*)alloc((size_t)3 * DH * DH * 2);       // 1.5 MB
  __bf16* hglob  = (__bf16*)alloc((size_t)2 * MROWS * DH * 2);    //  32 KB
  __bf16* rhglob = (__bf16*)alloc((size_t)MROWS * DH * 2);        //  16 KB
  __bf16* seq    = (__bf16*)alloc((size_t)BB * SS * DH * 2);      //  32 MB
  __bf16* Aout   = (__bf16*)alloc((size_t)3 * BB * SS * DH * 2);  //  96 MB

  pc_prep<<<1024, 256, 0, stream>>>(Wlm, Wz, Wr, Wh, wlm_bf, wgx_bf, wgh_bf,
                                    hglob, rhglob);
  // gemm1: 2048 M-tiles x 2 blocks (8 waves = 8 N-pairs each)
  pc_gemm1<<<4096, 256, 0, stream>>>(x, wlm_bf, seq);
  // gemm2: 2048 M-tiles x 6 blocks (8 waves = 8 N-pairs each)
  pc_gemm2<<<12288, 256, 0, stream>>>(seq, wgx_bf, Aout);
  // sequential scan: one 16-WG cluster, 4 waves per WG
  pc_scan<<<16, 128, 0, stream>>>(wgh_bf, Aout, hglob, rhglob);
  // classifier (final h is in hglob buffer 0 after 4096 steps)
  pc_head<<<1, 128, 0, stream>>>(hglob, Wc, out);
}
